// Layer_AGG_27470610825587
// MI455X (gfx1250) — matrix operand order repaired and checked by reading, not compiled, not run on hardware
//
#include <hip/hip_runtime.h>
#include <hip/hip_bf16.h>

typedef __attribute__((ext_vector_type(16))) __bf16 v16bf;
typedef __attribute__((ext_vector_type(8)))  float  v8f;

#define FEAT 128
#define KTOT 256

#if __has_builtin(__builtin_amdgcn_global_load_async_to_lds_b128) && \
    __has_builtin(__builtin_amdgcn_s_wait_asynccnt)
#define USE_ASYNC_LDS 1
typedef int i4 __attribute__((vector_size(16)));
typedef __attribute__((address_space(1))) i4 gi4;   // global int4
typedef __attribute__((address_space(3))) i4 li4;   // LDS int4
#else
#define USE_ASYNC_LDS 0
#endif

// ---------------------------------------------------------------- utilities
__global__ void zero_f(float* __restrict__ p, int n) {
    int i = blockIdx.x * 256 + threadIdx.x;
    if (i < n) p[i] = 0.0f;
}

// Build combined bf16 weight matrix Bt[o*256 + k] = (k<128 ? Wl[o,k] : Wr[o,k-128])
__global__ void prep_w(const float* __restrict__ Wl, const float* __restrict__ Wr,
                       __bf16* __restrict__ Bt) {
    int i = blockIdx.x * 256 + threadIdx.x;          // 0 .. 128*256-1
    if (i < FEAT * KTOT) {
        int o = i >> 8;
        int k = i & 255;
        float v = (k < FEAT) ? Wl[o * FEAT + k] : Wr[o * FEAT + (k - FEAT)];
        Bt[i] = (__bf16)v;
    }
}

// One wave per edge: atomic scatter-add of the 128-wide source row + degree count.
__global__ void scatter(const float* __restrict__ x, const int* __restrict__ src,
                        const int* __restrict__ dst, float* __restrict__ agg,
                        float* __restrict__ cnt, int nedges) {
    int e = blockIdx.x * 8 + (threadIdx.x >> 5);
    if (e >= nedges) return;
    int lane = threadIdx.x & 31;
    int s = src[e], d = dst[e];
    const float* xs = x   + (size_t)s * FEAT + lane * 4;
    float*       ad = agg + (size_t)d * FEAT + lane * 4;
#pragma unroll
    for (int j = 0; j < 4; ++j) unsafeAtomicAdd(&ad[j], xs[j]);
    if (lane == 0) unsafeAtomicAdd(&cnt[d], 1.0f);
}

__global__ void mean_div(float* __restrict__ agg, const float* __restrict__ cnt, int total) {
    int i = blockIdx.x * 256 + threadIdx.x;
    if (i < total) {
        float c = cnt[i >> 7];
        agg[i] = agg[i] / (c < 1.0f ? 1.0f : c);
    }
}

// pack 8 f32 (two float4) into bf16 vector lanes [base, base+8)
__device__ __forceinline__ void cvt8(v16bf& f, int base, float4 a, float4 b) {
    f[base + 0] = (__bf16)a.x; f[base + 1] = (__bf16)a.y;
    f[base + 2] = (__bf16)a.z; f[base + 3] = (__bf16)a.w;
    f[base + 4] = (__bf16)b.x; f[base + 5] = (__bf16)b.y;
    f[base + 6] = (__bf16)b.z; f[base + 7] = (__bf16)b.w;
}

// ------------------------------------------------------------- WMMA GEMM
// out[N,128] = [agg | self] (N x 256, bf16) @ Bt^T (+bias, optional relu)
// Bt layout: [o=0..127][k=0..255] bf16, staged whole in LDS (64 KB).
// Block: 256 threads = 8 waves. Wave w handles rows [blk*128 + w*16, +16), all 128 cols.
__global__ __launch_bounds__(256)
void sage_gemm(const float* __restrict__ Aagg, const float* __restrict__ Aself,
               const __bf16* __restrict__ Bt, const float* __restrict__ bias,
               float* __restrict__ Out, int nrows, int relu) {
    __shared__ __align__(32) __bf16 bsh[FEAT * KTOT];      // 64 KB

    int tid = threadIdx.x;
#if USE_ASYNC_LDS
    {   // async direct-to-LDS stage: 256 threads x 16 x 16B = 64 KB, ASYNCcnt tracked
        const char* g = (const char*)Bt;
        char*       l = (char*)bsh;
#pragma unroll
        for (int i = 0; i < 16; ++i) {
            int off = (tid + i * 256) * 16;
            __builtin_amdgcn_global_load_async_to_lds_b128(
                (gi4*)(g + off), (li4*)(l + off), 0, 0);
        }
        __builtin_amdgcn_s_wait_asynccnt(0);
    }
#else
    {   // fallback: dword copy through VGPRs
        const uint32_t* s = (const uint32_t*)Bt;
        uint32_t*       d = (uint32_t*)bsh;
        for (int i = tid; i < FEAT * KTOT / 2; i += 256) d[i] = s[i];
    }
#endif
    __syncthreads();

    int wave = tid >> 5;
    int lane = tid & 31;
    int l16  = lane & 15;
    int lhi  = lane >> 4;                                   // 0 or 1

    int rowBase = blockIdx.x * 128 + wave * 16;
    int arow    = rowBase + l16;
    arow = (arow < nrows) ? arow : (nrows - 1);             // clamp loads, guard stores

    const float4* aggRow  = (const float4*)(Aagg  + (size_t)arow * FEAT);
    const float4* selfRow = (const float4*)(Aself + (size_t)arow * FEAT);

    // 8 output tiles of 16x16 per wave; init accumulators with bias (depends on col only)
    v8f acc[8];
#pragma unroll
    for (int nt = 0; nt < 8; ++nt) {
        float b = bias[nt * 16 + l16];
#pragma unroll
        for (int r = 0; r < 8; ++r) acc[nt][r] = b;
    }

#pragma unroll
    for (int ks = 0; ks < 8; ++ks) {                        // K = 8 * 32 = 256
        int kbase = ks * 32;
        // ---- A fragment 16x32, k order shared with the B fragments:
        // contiguous: lane<16: K = kbase+{0..15}; lane>=16: K = kbase+{16..31}
        v16bf afrag;
        {
            int k0 = kbase + lhi * 16;                      // multiple of 16 -> 32B aligned
            const float4* s0 = (k0 < FEAT) ? (aggRow  + (k0 >> 2))
                                           : (selfRow + ((k0 - FEAT) >> 2));
            cvt8(afrag, 0, s0[0], s0[1]);
            int k1 = k0 + 8;
            const float4* s1 = (k1 < FEAT) ? (aggRow  + (k1 >> 2))
                                           : (selfRow + ((k1 - FEAT) >> 2));
            cvt8(afrag, 8, s1[0], s1[1]);
        }
        // ---- 8 B fragments from LDS: col n = nt*16+l16, K = kbase + lhi*16 + 0..15
#pragma unroll
        for (int nt = 0; nt < 8; ++nt) {
            const __bf16* bp = &bsh[(nt * 16 + l16) * KTOT + kbase + lhi * 16];
            v16bf bfrag = *(const v16bf*)bp;                // 32B aligned contiguous
            acc[nt] = __builtin_amdgcn_wmma_f32_16x16x32_bf16(
                false, afrag, false, bfrag, (short)0, acc[nt], false, false);
        }
    }

    // ---- store per ISA 32-bit C/D layout: VGPR r -> row rowBase + lhi*8 + r, col nt*16+l16
#pragma unroll
    for (int nt = 0; nt < 8; ++nt) {
        int col = nt * 16 + l16;
#pragma unroll
        for (int r = 0; r < 8; ++r) {
            int row = rowBase + lhi * 8 + r;
            if (row < nrows) {
                float v = acc[nt][r];
                if (relu) v = v > 0.0f ? v : 0.0f;
                Out[(size_t)row * FEAT + col] = v;
            }
        }
    }
}

// ------------------------------------------------------ gating + residual
// scores_i = <t_i[n,:], gw[i]> + gb[i]; alpha = softmax; out += a0*t0 + a1*t1
__global__ void gate_combine(const float* __restrict__ t0, const float* __restrict__ t1,
                             const float* __restrict__ gw, const float* __restrict__ gb,
                             float* __restrict__ out, int nnodes) {
    int n = blockIdx.x * 8 + (threadIdx.x >> 5);
    if (n >= nnodes) return;
    int lane = threadIdx.x & 31;
    const float* a = t0 + (size_t)n * FEAT + lane * 4;
    const float* b = t1 + (size_t)n * FEAT + lane * 4;
    float va[4], vb[4];
    float p0 = 0.0f, p1 = 0.0f;
#pragma unroll
    for (int j = 0; j < 4; ++j) {
        va[j] = a[j]; vb[j] = b[j];
        p0 += va[j] * gw[lane * 4 + j];
        p1 += vb[j] * gw[FEAT + lane * 4 + j];
    }
#pragma unroll
    for (int off = 16; off >= 1; off >>= 1) {
        p0 += __shfl_xor(p0, off, 32);
        p1 += __shfl_xor(p1, off, 32);
    }
    float s0 = p0 + gb[0], s1 = p1 + gb[1];
    float m  = fmaxf(s0, s1);
    float e0 = __expf(s0 - m), e1 = __expf(s1 - m);
    float inv = 1.0f / (e0 + e1);
    float a0 = e0 * inv, a1 = e1 * inv;
    float* o = out + (size_t)n * FEAT + lane * 4;
#pragma unroll
    for (int j = 0; j < 4; ++j) o[j] = o[j] + a0 * va[j] + a1 * vb[j];
}

// ---------------------------------------------------------------- driver
static void run_sage(const float* xin, const int* src, const int* dst,
                     const __bf16* Bt, const float* bias,
                     float* agg, float* cnt, float* outbuf,
                     int N, int E, int relu, hipStream_t stream) {
    int tot = N * FEAT;
    zero_f<<<(tot + 255) / 256, 256, 0, stream>>>(agg, tot);
    zero_f<<<(N + 255) / 256, 256, 0, stream>>>(cnt, N);
    scatter<<<(E + 7) / 8, 256, 0, stream>>>(xin, src, dst, agg, cnt, E);
    mean_div<<<(tot + 255) / 256, 256, 0, stream>>>(agg, cnt, tot);
    sage_gemm<<<(N + 127) / 128, 256, 0, stream>>>(agg, xin, Bt, bias, outbuf, N, relu);
}

extern "C" void kernel_launch(void* const* d_in, const int* in_sizes, int n_in,
                              void* d_out, int out_size, void* d_ws, size_t ws_size,
                              hipStream_t stream) {
    (void)n_in; (void)out_size; (void)ws_size;
    const float* x      = (const float*)d_in[0];
    const int*   e_main = (const int*)d_in[1];
    const int*   e_tree = (const int*)d_in[2];
    const float* cWl    = (const float*)d_in[3];
    const float* cbl    = (const float*)d_in[4];
    const float* cWr    = (const float*)d_in[5];
    const float* tWl    = (const float*)d_in[6];
    const float* tbl    = (const float*)d_in[7];
    const float* tWr    = (const float*)d_in[8];
    const float* gW     = (const float*)d_in[9];
    const float* gb     = (const float*)d_in[10];
    float* out = (float*)d_out;

    int N = in_sizes[0] / FEAT;
    int E = in_sizes[1] / 2;

    // workspace carve
    char* ws = (char*)d_ws;
    size_t fN = (size_t)N * FEAT * sizeof(float);
    float*  agg  = (float*)ws;  ws += fN;
    float*  bufA = (float*)ws;  ws += fN;          // h1, later tree branch 0
    float*  bufB = (float*)ws;  ws += fN;          // tree branch 1
    float*  cnt  = (float*)ws;  ws += ((size_t)N * sizeof(float) + 255) & ~(size_t)255;
    __bf16* Bt0  = (__bf16*)ws;                    // 4 x 128*256 bf16 = 256 KB
    __bf16* Bt1  = Bt0 + FEAT * KTOT;
    __bf16* Bt2  = Bt1 + FEAT * KTOT;
    __bf16* Bt3  = Bt2 + FEAT * KTOT;

    // fuse lin_l / lin_r weights into bf16 combined matrices
    prep_w<<<128, 256, 0, stream>>>(cWl,               cWr,               Bt0);
    prep_w<<<128, 256, 0, stream>>>(cWl + FEAT * FEAT, cWr + FEAT * FEAT, Bt1);
    prep_w<<<128, 256, 0, stream>>>(tWl,               tWr,               Bt2);
    prep_w<<<128, 256, 0, stream>>>(tWl + FEAT * FEAT, tWr + FEAT * FEAT, Bt3);

    // main chain: h1 = relu(sage(x)); out = sage(h1)   (written directly to d_out)
    run_sage(x,    e_main, e_main + E, Bt0, cbl,        agg, cnt, bufA, N, E, 1, stream);
    run_sage(bufA, e_main, e_main + E, Bt1, cbl + FEAT, agg, cnt, out,  N, E, 0, stream);

    // tree branches on original x, each with its own edge set
    run_sage(x, e_tree,         e_tree + E,     Bt2, tbl,        agg, cnt, bufA, N, E, 1, stream);
    run_sage(x, e_tree + 2 * E, e_tree + 3 * E, Bt3, tbl + FEAT, agg, cnt, bufB, N, E, 1, stream);

    // gated blend added onto the main-chain output (residual x + x_tree)
    gate_combine<<<(N + 7) / 8, 256, 0, stream>>>(bufA, bufB, gW, gb, out, N);
}
